// GATLayer_7645041787170
// MI455X (gfx1250) — compile-verified, matrix operand order checked by reading, not run on hardware
//
#include <hip/hip_runtime.h>

// ---------------------------------------------------------------------------
// GAT layer for MI455X (gfx1250, wave32):
//   ft = feat @ W          -> WMMA bf16 GEMM; B staged in LDS (async copy)
//   a1/a2 node attention   -> VALU dot pass
//   edge softmax           -> atomicMax (monotone-uint) + atomicAdd segments
//   scatter aggregation    -> float4 gathers + f32 atomics (HBM-bound)
//   ELU epilogue           -> in place on d_out
// ---------------------------------------------------------------------------

#define IN_DIM 256
#define HEADS  8
#define HDIM   32
#define ALPHA  0.2f

#define AS1 __attribute__((address_space(1)))
#define AS3 __attribute__((address_space(3)))

typedef __attribute__((ext_vector_type(16))) __bf16          v16bf;
typedef __attribute__((ext_vector_type(8)))  float           v8f;
typedef __attribute__((ext_vector_type(16))) unsigned short  v16u;
typedef int v4i __attribute__((vector_size(16)));            // matches builtin proto

__device__ __forceinline__ unsigned short f2bf(float f) {
    unsigned u = __builtin_bit_cast(unsigned, f);
    u += 0x7FFFu + ((u >> 16) & 1u);        // round-to-nearest-even
    return (unsigned short)(u >> 16);
}

// Monotonic float <-> uint encoding for atomicMax on signed floats.
__device__ __forceinline__ unsigned fenc(float f) {
    unsigned u = __builtin_bit_cast(unsigned, f);
    return (u >> 31) ? ~u : (u | 0x80000000u);
}
__device__ __forceinline__ float fdec(unsigned e) {
    unsigned u = (e & 0x80000000u) ? (e ^ 0x80000000u) : ~e;
    return __builtin_bit_cast(float, u);
}

// K index inside a 16x32 bf16 WMMA fragment for element j of lane (half,l16).
__device__ __forceinline__ int frag_k(int j, int half) {
    int v = j >> 1, p = j & 1;
    return ((v < 4) ? 0 : 16) + ((v & 3) * 2) + p + half * 8;
}

// ---------------------------------------------------------------------------
// Pre-swizzle W (256x256 f32, row major) into fragment-major bf16, grouped so
// each 64-column group (g = ntile>>2) is one contiguous 32KB run:
//   Wswz[g][ks][t4][lane][j]   (g:4, ks:8, t4:4, lane:32, j:16 bf16)
// ---------------------------------------------------------------------------
__global__ void k_wswz(const float* __restrict__ W, unsigned short* __restrict__ Wswz) {
    int tid  = blockIdx.x * blockDim.x + threadIdx.x;   // 0..4095
    if (tid >= 8 * 16 * 32) return;
    int lane = tid & 31;
    int nt   = (tid >> 5) & 15;
    int ks   = tid >> 9;
    int half = lane >> 4, l16 = lane & 15;
    int g = nt >> 2, t4 = nt & 3;
    unsigned short* dst = Wswz + ((((size_t)g * 8 + ks) * 4 + t4) * 32 + lane) * 16;
#pragma unroll
    for (int j = 0; j < 16; ++j) {
        int k = frag_k(j, half);
        dst[j] = f2bf(W[(size_t)(ks * 32 + k) * IN_DIM + nt * 16 + l16]);
    }
}

// ---------------------------------------------------------------------------
// Projection GEMM: ft[N,256] = feat[N,256] @ W[256,256] via WMMA bf16.
// 256 threads = 8 waves; wave owns 16(M) x 64(N); block covers 128 rows.
// B fragments for the whole K loop staged into 32KB LDS (async copy).
// ---------------------------------------------------------------------------
__global__ __launch_bounds__(256)
void k_proj_wmma(const float* __restrict__ feat,
                 const unsigned short* __restrict__ Wswz,
                 float* __restrict__ ft, int nrows) {
    __shared__ __attribute__((aligned(32))) unsigned short ldsW[16384];  // 32KB

    const int lane = threadIdx.x & 31;
    const int wave = threadIdx.x >> 5;
    const int half = lane >> 4;
    const int l16  = lane & 15;
    const int m0   = blockIdx.x * 128 + wave * 16;
    const int n0   = blockIdx.y * 64;

    // ---- Stage this column group's B fragments (32KB) into LDS ----
    const unsigned short* gsrc = Wswz + (size_t)blockIdx.y * 16384;
#if __has_builtin(__builtin_amdgcn_global_load_async_to_lds_b128)
    {
        AS1 v4i* gb = (AS1 v4i*)gsrc;
        AS3 v4i* lb = (AS3 v4i*)ldsW;
#pragma unroll
        for (int i = 0; i < 8; ++i) {
            int idx = i * 256 + threadIdx.x;            // 16B chunk index
            __builtin_amdgcn_global_load_async_to_lds_b128(gb + idx, lb + idx, 0, 0);
        }
    }
#if __has_builtin(__builtin_amdgcn_s_wait_asynccnt)
    __builtin_amdgcn_s_wait_asynccnt(0);
#else
    asm volatile("s_wait_asynccnt 0x0" ::: "memory");
#endif
#else
#pragma unroll
    for (int i = 0; i < 8; ++i) {
        int idx = i * 256 + threadIdx.x;
        ((float4*)ldsW)[idx] = ((const float4*)gsrc)[idx];
    }
#endif
    __syncthreads();

    v8f acc0 = {}, acc1 = {}, acc2 = {}, acc3 = {};

    int row = m0 + l16;
    if (row >= nrows) row = nrows - 1;              // clamp; stores are guarded
    const float* arow = feat + (size_t)row * IN_DIM + half * 8;

    const v16u* lfrag = (const v16u*)ldsW;          // [ks][t4][lane] 32B frags

#pragma unroll
    for (int ks = 0; ks < 8; ++ks) {
        // ---- A fragment: two contiguous 8-float runs -> 16 bf16 ----
        const float* ap = arow + ks * 32;
        float4 fa = *(const float4*)(ap + 0);
        float4 fb = *(const float4*)(ap + 4);
        float4 fc = *(const float4*)(ap + 16);
        float4 fd = *(const float4*)(ap + 20);
        v16u au;
        au[0]  = f2bf(fa.x); au[1]  = f2bf(fa.y); au[2]  = f2bf(fa.z); au[3]  = f2bf(fa.w);
        au[4]  = f2bf(fb.x); au[5]  = f2bf(fb.y); au[6]  = f2bf(fb.z); au[7]  = f2bf(fb.w);
        au[8]  = f2bf(fc.x); au[9]  = f2bf(fc.y); au[10] = f2bf(fc.z); au[11] = f2bf(fc.w);
        au[12] = f2bf(fd.x); au[13] = f2bf(fd.y); au[14] = f2bf(fd.z); au[15] = f2bf(fd.w);
        v16bf aF = __builtin_bit_cast(v16bf, au);

        // ---- B fragments from LDS: one 32B ds read per lane each ----
        v16bf b0 = __builtin_bit_cast(v16bf, lfrag[(ks * 4 + 0) * 32 + lane]);
        v16bf b1 = __builtin_bit_cast(v16bf, lfrag[(ks * 4 + 1) * 32 + lane]);
        v16bf b2 = __builtin_bit_cast(v16bf, lfrag[(ks * 4 + 2) * 32 + lane]);
        v16bf b3 = __builtin_bit_cast(v16bf, lfrag[(ks * 4 + 3) * 32 + lane]);

        acc0 = __builtin_amdgcn_wmma_f32_16x16x32_bf16(false, aF, false, b0, (short)0, acc0, false, false);
        acc1 = __builtin_amdgcn_wmma_f32_16x16x32_bf16(false, aF, false, b1, (short)0, acc1, false, false);
        acc2 = __builtin_amdgcn_wmma_f32_16x16x32_bf16(false, aF, false, b2, (short)0, acc2, false, false);
        acc3 = __builtin_amdgcn_wmma_f32_16x16x32_bf16(false, aF, false, b3, (short)0, acc3, false, false);
    }

    // ---- Store C: VGPR r -> row m0 + half*8 + r, col = n0 + t*16 + l16 ----
    v8f accs[4] = {acc0, acc1, acc2, acc3};
    if (m0 + 16 <= nrows) {
        // fast path: whole wave tile in range, unguarded coalesced stores
        float* cp = ft + (size_t)(m0 + half * 8) * IN_DIM + n0 + l16;
#pragma unroll
        for (int t = 0; t < 4; ++t) {
#pragma unroll
            for (int r = 0; r < 8; ++r)
                cp[(size_t)r * IN_DIM + t * 16] = accs[t][r];
        }
    } else {
#pragma unroll
        for (int t = 0; t < 4; ++t) {
            int col = n0 + t * 16 + l16;
#pragma unroll
            for (int r = 0; r < 8; ++r) {
                int m = m0 + half * 8 + r;
                if (m < nrows) ft[(size_t)m * IN_DIM + col] = accs[t][r];
            }
        }
    }
}

// ---------------------------------------------------------------------------
// a1[n,h] = <ft[n,h,:], attn_l[h,:]> ; a2 likewise. One thread per (n,h).
// ---------------------------------------------------------------------------
__global__ void k_node_attn(const float* __restrict__ ft,
                            const float* __restrict__ attn_l,
                            const float* __restrict__ attn_r,
                            float* __restrict__ a1, float* __restrict__ a2, int nH) {
    int tid = blockIdx.x * blockDim.x + threadIdx.x;
    if (tid >= nH) return;
    int n = tid >> 3, h = tid & 7;
    const float4* v  = (const float4*)(ft + (size_t)n * IN_DIM + h * HDIM);
    const float4* al = (const float4*)(attn_l + h * HDIM);
    const float4* ar = (const float4*)(attn_r + h * HDIM);
    float s1 = 0.f, s2 = 0.f;
#pragma unroll
    for (int i = 0; i < 8; ++i) {
        float4 x = v[i], l = al[i], r = ar[i];
        s1 += x.x * l.x + x.y * l.y + x.z * l.z + x.w * l.w;
        s2 += x.x * r.x + x.y * r.y + x.z * r.z + x.w * r.w;
    }
    a1[tid] = s1; a2[tid] = s2;
}

// ---------------------------------------------------------------------------
// Edge pass 1: segment max of LeakyReLU logits (monotone-uint atomicMax).
// ---------------------------------------------------------------------------
__global__ void k_edge_max(const int* __restrict__ src, const int* __restrict__ dst,
                           const float* __restrict__ a1, const float* __restrict__ a2,
                           unsigned* __restrict__ maxb, int EH) {
    int tid = blockIdx.x * blockDim.x + threadIdx.x;
    if (tid >= EH) return;
    int e = tid >> 3, h = tid & 7;
    int s = src[e], d = dst[e];
    float x = a1[s * HEADS + h] + a2[d * HEADS + h];
    x = (x > 0.f) ? x : ALPHA * x;
    atomicMax(&maxb[d * HEADS + h], fenc(x));
}

// ---------------------------------------------------------------------------
// Edge pass 2: ex = exp(e - max); store per-edge ex + segment sum.
// ---------------------------------------------------------------------------
__global__ void k_edge_sum(const int* __restrict__ src, const int* __restrict__ dst,
                           const float* __restrict__ a1, const float* __restrict__ a2,
                           const unsigned* __restrict__ maxb,
                           float* __restrict__ exbuf, float* __restrict__ sumb, int EH) {
    int tid = blockIdx.x * blockDim.x + threadIdx.x;
    if (tid >= EH) return;
    int e = tid >> 3, h = tid & 7;
    int s = src[e], d = dst[e];
    float x = a1[s * HEADS + h] + a2[d * HEADS + h];
    x = (x > 0.f) ? x : ALPHA * x;
    float ex = __expf(x - fdec(maxb[d * HEADS + h]));
    exbuf[tid] = ex;
    atomicAdd(&sumb[d * HEADS + h], ex);
}

// ---------------------------------------------------------------------------
// Edge pass 3: out[dst,h,:] += alpha * ft[src,h,:]  (float4 gather + atomics)
// ---------------------------------------------------------------------------
__global__ void k_aggregate(const int* __restrict__ src, const int* __restrict__ dst,
                            const float* __restrict__ exbuf, const float* __restrict__ sumb,
                            const float* __restrict__ ft, float* __restrict__ out, int EH) {
    int tid = blockIdx.x * blockDim.x + threadIdx.x;
    if (tid >= EH) return;
    int e = tid >> 3, h = tid & 7;
    int s = src[e], d = dst[e];
    float alpha = exbuf[tid] / sumb[d * HEADS + h];
    const float4* fv = (const float4*)(ft + (size_t)s * IN_DIM + h * HDIM);
    float* o = out + (size_t)d * IN_DIM + h * HDIM;
#pragma unroll
    for (int i = 0; i < 8; ++i) {
        float4 v = fv[i];
        atomicAdd(o + i * 4 + 0, alpha * v.x);
        atomicAdd(o + i * 4 + 1, alpha * v.y);
        atomicAdd(o + i * 4 + 2, alpha * v.z);
        atomicAdd(o + i * 4 + 3, alpha * v.w);
    }
}

// ---------------------------------------------------------------------------
// ELU epilogue, in place, float4 vectorized.
// ---------------------------------------------------------------------------
__global__ void k_elu(float* __restrict__ out, int n4) {
    int tid = blockIdx.x * blockDim.x + threadIdx.x;
    if (tid >= n4) return;
    float4 v = ((const float4*)out)[tid];
    v.x = (v.x > 0.f) ? v.x : (__expf(v.x) - 1.f);
    v.y = (v.y > 0.f) ? v.y : (__expf(v.y) - 1.f);
    v.z = (v.z > 0.f) ? v.z : (__expf(v.z) - 1.f);
    v.w = (v.w > 0.f) ? v.w : (__expf(v.w) - 1.f);
    ((float4*)out)[tid] = v;
}

// ---------------------------------------------------------------------------
extern "C" void kernel_launch(void* const* d_in, const int* in_sizes, int n_in,
                              void* d_out, int out_size, void* d_ws, size_t ws_size,
                              hipStream_t stream) {
    const float* feat   = (const float*)d_in[0];
    const int*   src    = (const int*)d_in[1];
    const int*   dst    = (const int*)d_in[2];
    const float* W      = (const float*)d_in[3];
    const float* attn_l = (const float*)d_in[4];
    const float* attn_r = (const float*)d_in[5];

    const int N  = in_sizes[0] / IN_DIM;
    const int E  = in_sizes[1];
    const int NH = N * HEADS;
    const int EH = E * HEADS;

    // Workspace layout
    float*          ft    = (float*)d_ws;                       // N*256
    float*          a1    = ft + (size_t)N * IN_DIM;            // N*8
    float*          a2    = a1 + (size_t)NH;                    // N*8
    unsigned*       maxb  = (unsigned*)(a2 + (size_t)NH);       // N*8
    float*          sumb  = (float*)(maxb + (size_t)NH);        // N*8
    float*          exbuf = sumb + (size_t)NH;                  // E*8
    unsigned short* wswz  = (unsigned short*)(exbuf + (size_t)EH); // 64K bf16

    float* out = (float*)d_out;

    // Zero-init accumulation buffers (graph-capture-safe memset nodes)
    (void)hipMemsetAsync(out,  0, (size_t)N * IN_DIM * sizeof(float), stream);
    (void)hipMemsetAsync(maxb, 0, (size_t)NH * sizeof(unsigned), stream);
    (void)hipMemsetAsync(sumb, 0, (size_t)NH * sizeof(float), stream);

    // 1) Swizzle W into WMMA B-fragment layout (bf16)
    k_wswz<<<16, 256, 0, stream>>>(W, wswz);

    // 2) Projection GEMM via WMMA
    dim3 ggrid((N + 127) / 128, 4);
    k_proj_wmma<<<ggrid, 256, 0, stream>>>(feat, wswz, ft, N);

    // 3) Node attention terms
    k_node_attn<<<(NH + 255) / 256, 256, 0, stream>>>(ft, attn_l, attn_r, a1, a2, NH);

    // 4) Edge softmax + aggregation
    int eblocks = (EH + 255) / 256;
    k_edge_max<<<eblocks, 256, 0, stream>>>(src, dst, a1, a2, maxb, EH);
    k_edge_sum<<<eblocks, 256, 0, stream>>>(src, dst, a1, a2, maxb, exbuf, sumb, EH);
    k_aggregate<<<eblocks, 256, 0, stream>>>(src, dst, exbuf, sumb, ft, out, EH);

    // 5) ELU epilogue
    int n4 = N * IN_DIM / 4;
    k_elu<<<(n4 + 255) / 256, 256, 0, stream>>>(out, n4);
}
